// FallDetectionSGAT_38465727103396
// MI455X (gfx1250) — matrix operand order repaired and checked by reading, not compiled, not run on hardware
//
#include <hip/hip_runtime.h>
#include <hip/hip_bf16.h>

typedef __attribute__((ext_vector_type(16))) _Float16 v16h;
typedef __attribute__((ext_vector_type(8)))  float    v8f;

#define NN 131072
#define EE 2097152
#define GG 512
#define ETOT (EE + NN)

// ---------------------------------------------------------------------------
// utility
// ---------------------------------------------------------------------------
__global__ void fill_f32(float* p, float v, long long count) {
  long long i = (long long)blockIdx.x * blockDim.x + threadIdx.x;
  if (i < count) p[i] = v;
}

__device__ inline void atomicMaxF(float* addr, float value) {
  // sign-split trick: IEEE float order == int order for >=0, reversed uint order for <0
  if (value >= 0.0f) {
    atomicMax((int*)addr, __float_as_int(value));
  } else {
    atomicMin((unsigned int*)addr, __float_as_uint(value));
  }
}

// ---------------------------------------------------------------------------
// self-loop attrs: cnt[d] += 1, lsum[d] += edge_attr[e]; then lsum /= max(cnt,1)
// ---------------------------------------------------------------------------
__global__ void loop_accum(const int* __restrict__ ei, const float* __restrict__ eattr,
                           float* __restrict__ cnt, float* __restrict__ lsum) {
  long long e = (long long)blockIdx.x * blockDim.x + threadIdx.x;
  if (e >= EE) return;
  int d = ei[EE + e];
  atomicAdd(&cnt[d], 1.0f);
  atomicAdd(&lsum[(long long)d * 3 + 0], eattr[e * 3 + 0]);
  atomicAdd(&lsum[(long long)d * 3 + 1], eattr[e * 3 + 1]);
  atomicAdd(&lsum[(long long)d * 3 + 2], eattr[e * 3 + 2]);
}

__global__ void loop_div(float* __restrict__ lsum, const float* __restrict__ cnt) {
  long long t = (long long)blockIdx.x * blockDim.x + threadIdx.x;
  if (t >= (long long)NN * 3) return;
  int node = (int)(t / 3);
  lsum[t] /= fmaxf(cnt[node], 1.0f);
}

// ---------------------------------------------------------------------------
// WMMA GEMM: OUT[n, COUT] = X[n, K] @ W[K, COUT] + bias  (f32 io, f16 MAC, f32 acc)
// one wave -> 16 rows x 16 cols; block has COUT/16 waves; grid.x = n/16
// ---------------------------------------------------------------------------
template <int K, int COUT>
__global__ void wmma_gemm_bias(const float* __restrict__ X, const float* __restrict__ W,
                               const float* __restrict__ bias, float* __restrict__ out, int n) {
  int wave = threadIdx.x >> 5;
  int lane = threadIdx.x & 31;
  int lo = lane & 15;
  int hi = lane >> 4;
  int m0 = blockIdx.x << 4;
  int c0 = wave << 4;
  int row = m0 + lo;

  v8f acc = {};
  for (int kb = 0; kb < K; kb += 32) {
    v16h a, b;
    // A tile 16x32 f16: lane(0-15)=row M, hi selects K-half per ISA 7.12.2
    const float* xrow = X + (long long)row * K + kb;
#pragma unroll
    for (int v = 0; v < 8; v++) {
      int grp = v >> 2, vv = v & 3;
      int k = grp * 16 + hi * 8 + 2 * vv;
      a[2 * v]     = (_Float16)xrow[k];
      a[2 * v + 1] = (_Float16)xrow[k + 1];
    }
    // B tile 32x16 f16: lane(0-15)=col N, same K order as A (hi selects K-half)
    const float* wb = W + (long long)kb * COUT + c0 + lo;
#pragma unroll
    for (int v = 0; v < 8; v++) {
      int k = (v >> 2) * 16 + hi * 8 + 2 * (v & 3);
      b[2 * v]     = (_Float16)wb[(long long)k * COUT];
      b[2 * v + 1] = (_Float16)wb[(long long)(k + 1) * COUT];
    }
    acc = __builtin_amdgcn_wmma_f32_16x16x32_f16(false, a, false, b, (short)0, acc,
                                                 false, false);
  }

  float bval = bias[c0 + lo];
#pragma unroll
  for (int r = 0; r < 8; r++) {
    int m = m0 + hi * 8 + r;  // C/D layout: VGPR r -> M=r (lanes 0-15) / M=8+r (lanes 16-31)
    out[(long long)m * COUT + c0 + lo] = acc[r] + bval;
  }
}

// ---------------------------------------------------------------------------
// edge pass A: logits + segment max  (one thread per (edge, head); H==2)
// b128 gathers of xl[src]/xr[dst]; weight/att vectors are broadcast (L2-resident)
// ---------------------------------------------------------------------------
template <int C, int HC>
__global__ void edge_logits(const float* __restrict__ xl, const float* __restrict__ xr,
                            const float* __restrict__ eattr, const float* __restrict__ lattr,
                            const int* __restrict__ ei, const float* __restrict__ we,
                            const float* __restrict__ att, float* __restrict__ elog,
                            float* __restrict__ mx) {
  long long t = (long long)blockIdx.x * blockDim.x + threadIdx.x;
  if (t >= (long long)ETOT * 2) return;
  long long e = t >> 1;
  int h = (int)(t & 1);
  int s, d;
  const float* ea;
  if (e < EE) {
    s = ei[e]; d = ei[EE + e]; ea = eattr + e * 3;
  } else {
    s = d = (int)(e - EE);     ea = lattr + (e - EE) * 3;
  }
  float a0 = ea[0], a1 = ea[1], a2 = ea[2];
  const float4* xls = (const float4*)(xl + (long long)s * HC + h * C);
  const float4* xrd = (const float4*)(xr + (long long)d * HC + h * C);
  const float4* w0 = (const float4*)(we + h * C);
  const float4* w1 = (const float4*)(we + HC + h * C);
  const float4* w2 = (const float4*)(we + 2 * HC + h * C);
  const float4* at = (const float4*)(att + h * C);
  float logit = 0.0f;
#pragma unroll
  for (int q = 0; q < C / 4; q++) {
    float4 vl = xls[q], vr = xrd[q];
    float4 e0 = w0[q], e1 = w1[q], e2 = w2[q], aa = at[q];
    float v0 = vl.x + vr.x + a0 * e0.x + a1 * e1.x + a2 * e2.x;
    float v1 = vl.y + vr.y + a0 * e0.y + a1 * e1.y + a2 * e2.y;
    float v2 = vl.z + vr.z + a0 * e0.z + a1 * e1.z + a2 * e2.z;
    float v3 = vl.w + vr.w + a0 * e0.w + a1 * e1.w + a2 * e2.w;
    v0 = v0 > 0.0f ? v0 : 0.2f * v0;  // leaky_relu 0.2
    v1 = v1 > 0.0f ? v1 : 0.2f * v1;
    v2 = v2 > 0.0f ? v2 : 0.2f * v2;
    v3 = v3 > 0.0f ? v3 : 0.2f * v3;
    logit += v0 * aa.x + v1 * aa.y + v2 * aa.z + v3 * aa.w;
  }
  elog[e * 2 + h] = logit;
  atomicMaxF(&mx[(long long)d * 2 + h], logit);
}

// ---------------------------------------------------------------------------
// edge pass B: exp(logit - max) + segment sum (overwrites elog with exp)
// ---------------------------------------------------------------------------
__global__ void edge_exp(const int* __restrict__ ei, float* __restrict__ elog,
                         const float* __restrict__ mx, float* __restrict__ sm) {
  long long t = (long long)blockIdx.x * blockDim.x + threadIdx.x;
  if (t >= (long long)ETOT * 2) return;
  long long e = t >> 1;
  int h = (int)(t & 1);
  int d = (e < EE) ? ei[EE + e] : (int)(e - EE);
  float ex = expf(elog[t] - mx[(long long)d * 2 + h]);
  elog[t] = ex;
  atomicAdd(&sm[(long long)d * 2 + h], ex);
}

// ---------------------------------------------------------------------------
// edge pass C: agg[d] += alpha * xl[src]
// one thread per (edge, head): index/alpha loaded once, xl gathered as b128,
// C scalar f32 atomics (irreducible)
// ---------------------------------------------------------------------------
template <int C, int HC>
__global__ void edge_aggregate(const int* __restrict__ ei, const float* __restrict__ elog,
                               const float* __restrict__ sm, const float* __restrict__ xl,
                               float* __restrict__ agg) {
  long long t = (long long)blockIdx.x * blockDim.x + threadIdx.x;
  if (t >= (long long)ETOT * 2) return;
  long long e = t >> 1;
  int h = (int)(t & 1);
  int s, d;
  if (e < EE) { s = ei[e]; d = ei[EE + e]; }
  else        { s = d = (int)(e - EE); }
  float alpha = elog[e * 2 + h] / (sm[(long long)d * 2 + h] + 1e-16f);
  const float4* src4 = (const float4*)(xl + (long long)s * HC + h * C);
  float* dst = agg + (long long)d * HC + h * C;
#pragma unroll
  for (int q = 0; q < C / 4; q++) {
    float4 v = src4[q];
    atomicAdd(&dst[4 * q + 0], alpha * v.x);
    atomicAdd(&dst[4 * q + 1], alpha * v.y);
    atomicAdd(&dst[4 * q + 2], alpha * v.z);
    atomicAdd(&dst[4 * q + 3], alpha * v.w);
  }
}

// ---------------------------------------------------------------------------
// head-mean over H=2 + bias  -> pre-BN features h[n, C]
// ---------------------------------------------------------------------------
template <int C>
__global__ void head_mean_bias(const float* __restrict__ agg, const float* __restrict__ bias,
                               float* __restrict__ h) {
  long long t = (long long)blockIdx.x * blockDim.x + threadIdx.x;
  if (t >= (long long)NN * C) return;
  long long node = t / C;
  int c = (int)(t % C);
  h[t] = 0.5f * (agg[node * 2 * C + c] + agg[node * 2 * C + C + c]) + bias[c];
}

// ---------------------------------------------------------------------------
// BN stats: acc[0..C) = sum, acc[C..2C) = sumsq. blockDim=256 (multiple of C),
// gridDim*256 stride keeps each thread on one fixed channel.
// ---------------------------------------------------------------------------
template <int C>
__global__ void bn_stats(const float* __restrict__ x, int n, float* __restrict__ acc) {
  __shared__ float s1[256], s2[256];
  int tid = threadIdx.x;
  float a = 0.0f, b = 0.0f;
  long long total = (long long)n * C;
  long long stride = (long long)gridDim.x * blockDim.x;
  for (long long i = (long long)blockIdx.x * blockDim.x + tid; i < total; i += stride) {
    float v = x[i];
    a += v;
    b += v * v;
  }
  s1[tid] = a; s2[tid] = b;
  __syncthreads();
  for (int s = 128; s >= C; s >>= 1) {
    if (tid < s) { s1[tid] += s1[tid + s]; s2[tid] += s2[tid + s]; }
    __syncthreads();
  }
  if (tid < C) {
    atomicAdd(&acc[tid], s1[tid]);
    atomicAdd(&acc[C + tid], s2[tid]);
  }
}

template <int C>
__global__ void bn_apply_elu(float* __restrict__ x, int n, const float* __restrict__ acc,
                             const float* __restrict__ g, const float* __restrict__ be) {
  long long t = (long long)blockIdx.x * blockDim.x + threadIdx.x;
  if (t >= (long long)n * C) return;
  int c = (int)(t % C);
  float inv_n = 1.0f / (float)n;
  float mu = acc[c] * inv_n;
  float var = acc[C + c] * inv_n - mu * mu;
  float rs = rsqrtf(var + 1e-5f);
  float v = (x[t] - mu) * rs * g[c] + be[c];
  x[t] = v > 0.0f ? v : (expf(v) - 1.0f);  // elu
}

// ---------------------------------------------------------------------------
// global mean pool (sums + counts; mean applied in head)
// ---------------------------------------------------------------------------
template <int C>
__global__ void pool_kernel(const float* __restrict__ h, const int* __restrict__ batch,
                            float* __restrict__ pool, float* __restrict__ gcnt) {
  long long t = (long long)blockIdx.x * blockDim.x + threadIdx.x;
  if (t >= (long long)NN * C) return;
  int node = (int)(t / C);
  int c = (int)(t % C);
  int g = batch[node];
  atomicAdd(&pool[(long long)g * C + c], h[t]);
  if (c == 0) atomicAdd(&gcnt[g], 1.0f);
}

// ---------------------------------------------------------------------------
// head: z = (pool/cnt) @ fc1w + fc1b       (G x 64) @ (64 x 32)
// ---------------------------------------------------------------------------
__global__ void head_fc1(const float* __restrict__ pool, const float* __restrict__ gcnt,
                         const float* __restrict__ fc1w, const float* __restrict__ fc1b,
                         float* __restrict__ z) {
  int t = blockIdx.x * blockDim.x + threadIdx.x;
  if (t >= GG * 32) return;
  int g = t >> 5, j = t & 31;
  float inv = 1.0f / fmaxf(gcnt[g], 1.0f);
  float acc = fc1b[j];
#pragma unroll 8
  for (int c = 0; c < 64; c++) acc += pool[g * 64 + c] * inv * fc1w[c * 32 + j];
  z[t] = acc;
}

// head: BN(z)+elu, then @ fc2w + fc2b, sigmoid
__global__ void head_out(const float* __restrict__ z, const float* __restrict__ acc,
                         const float* __restrict__ g3, const float* __restrict__ be3,
                         const float* __restrict__ fc2w, const float* __restrict__ fc2b,
                         float* __restrict__ out) {
  int g = blockIdx.x * blockDim.x + threadIdx.x;
  if (g >= GG) return;
  float s = fc2b[0];
  const float inv_g = 1.0f / (float)GG;
#pragma unroll 8
  for (int j = 0; j < 32; j++) {
    float mu = acc[j] * inv_g;
    float var = acc[32 + j] * inv_g - mu * mu;
    float v = (z[g * 32 + j] - mu) * rsqrtf(var + 1e-5f) * g3[j] + be3[j];
    v = v > 0.0f ? v : (expf(v) - 1.0f);
    s += v * fc2w[j];
  }
  out[g] = 1.0f / (1.0f + expf(-s));
}

// ---------------------------------------------------------------------------
// launch
// ---------------------------------------------------------------------------
static inline unsigned nblk(long long total, int bs) {
  return (unsigned)((total + bs - 1) / bs);
}

extern "C" void kernel_launch(void* const* d_in, const int* in_sizes, int n_in,
                              void* d_out, int out_size, void* d_ws, size_t ws_size,
                              hipStream_t stream) {
  const float* x     = (const float*)d_in[0];
  const float* eattr = (const float*)d_in[1];
  const float* w1l = (const float*)d_in[2];  const float* b1l = (const float*)d_in[3];
  const float* w1r = (const float*)d_in[4];  const float* b1r = (const float*)d_in[5];
  const float* w1e = (const float*)d_in[6];  const float* att1 = (const float*)d_in[7];
  const float* bias1 = (const float*)d_in[8];
  const float* g1 = (const float*)d_in[9];   const float* be1 = (const float*)d_in[10];
  const float* w2l = (const float*)d_in[11]; const float* b2l = (const float*)d_in[12];
  const float* w2r = (const float*)d_in[13]; const float* b2r = (const float*)d_in[14];
  const float* w2e = (const float*)d_in[15]; const float* att2 = (const float*)d_in[16];
  const float* bias2 = (const float*)d_in[17];
  const float* g2 = (const float*)d_in[18];  const float* be2 = (const float*)d_in[19];
  const float* fc1w = (const float*)d_in[20]; const float* fc1b = (const float*)d_in[21];
  const float* g3 = (const float*)d_in[22];  const float* be3 = (const float*)d_in[23];
  const float* fc2w = (const float*)d_in[24]; const float* fc2b = (const float*)d_in[25];
  const int* ei    = (const int*)d_in[26];
  const int* batch = (const int*)d_in[27];
  float* out = (float*)d_out;

  // ------ workspace layout (floats) ------
  float* ws = (float*)d_ws;
  const long long N = NN;
  float* cnt   = ws;                 // N
  float* lattr = cnt + N;            // 3N
  float* mx    = lattr + 3 * N;      // 2N
  float* sm    = mx + 2 * N;         // 2N
  float* h1    = sm + 2 * N;         // 32N
  float* h2    = h1 + 32 * N;        // 64N
  float* elog  = h2 + 64 * N;        // 2*ETOT
  float* small = elog + 2LL * ETOT;
  float* pool  = small;              // G*64
  float* gcnt  = pool + GG * 64;     // G
  float* z     = gcnt + GG;          // G*32
  float* bnacc = z + GG * 32;        // 256
  float* big   = bnacc + 256;        // 384N reused region
  // phase 1
  float* xl1 = big;                  // 64N
  float* xr1 = big + 64 * N;         // 64N
  float* agg1 = big + 128 * N;       // 64N
  // phase 2 (overlays phase-1 region)
  float* xl2 = big;                  // 128N
  float* xr2 = big + 128 * N;        // 128N
  float* agg2 = big + 256 * N;       // 128N

  const int BS = 256;

  // ---- self-loop edge attrs ----
  hipMemsetAsync(cnt, 0, N * sizeof(float), stream);
  hipMemsetAsync(lattr, 0, 3 * N * sizeof(float), stream);
  loop_accum<<<nblk(EE, BS), BS, 0, stream>>>(ei, eattr, cnt, lattr);
  loop_div<<<nblk(3 * N, BS), BS, 0, stream>>>(lattr, cnt);

  // ---- layer 1 projections (WMMA) ----
  wmma_gemm_bias<64, 64><<<NN / 16, 128, 0, stream>>>(x, w1l, b1l, xl1, NN);
  wmma_gemm_bias<64, 64><<<NN / 16, 128, 0, stream>>>(x, w1r, b1r, xr1, NN);

  // ---- layer 1 attention ----
  fill_f32<<<nblk(2 * N, BS), BS, 0, stream>>>(mx, -__builtin_inff(), 2 * N);
  hipMemsetAsync(sm, 0, 2 * N * sizeof(float), stream);
  edge_logits<32, 64><<<nblk(2LL * ETOT, BS), BS, 0, stream>>>(xl1, xr1, eattr, lattr, ei,
                                                               w1e, att1, elog, mx);
  edge_exp<<<nblk(2LL * ETOT, BS), BS, 0, stream>>>(ei, elog, mx, sm);
  hipMemsetAsync(agg1, 0, 64 * N * sizeof(float), stream);
  edge_aggregate<32, 64><<<nblk(2LL * ETOT, BS), BS, 0, stream>>>(ei, elog, sm, xl1, agg1);
  head_mean_bias<32><<<nblk(32 * N, BS), BS, 0, stream>>>(agg1, bias1, h1);
  hipMemsetAsync(bnacc, 0, 64 * sizeof(float), stream);
  bn_stats<32><<<512, 256, 0, stream>>>(h1, NN, bnacc);
  bn_apply_elu<32><<<nblk(32 * N, BS), BS, 0, stream>>>(h1, NN, bnacc, g1, be1);

  // ---- layer 2 projections (WMMA) ----
  wmma_gemm_bias<32, 128><<<NN / 16, 256, 0, stream>>>(h1, w2l, b2l, xl2, NN);
  wmma_gemm_bias<32, 128><<<NN / 16, 256, 0, stream>>>(h1, w2r, b2r, xr2, NN);

  // ---- layer 2 attention ----
  fill_f32<<<nblk(2 * N, BS), BS, 0, stream>>>(mx, -__builtin_inff(), 2 * N);
  hipMemsetAsync(sm, 0, 2 * N * sizeof(float), stream);
  edge_logits<64, 128><<<nblk(2LL * ETOT, BS), BS, 0, stream>>>(xl2, xr2, eattr, lattr, ei,
                                                                w2e, att2, elog, mx);
  edge_exp<<<nblk(2LL * ETOT, BS), BS, 0, stream>>>(ei, elog, mx, sm);
  hipMemsetAsync(agg2, 0, 128 * N * sizeof(float), stream);
  edge_aggregate<64, 128><<<nblk(2LL * ETOT, BS), BS, 0, stream>>>(ei, elog, sm, xl2, agg2);
  head_mean_bias<64><<<nblk(64 * N, BS), BS, 0, stream>>>(agg2, bias2, h2);
  hipMemsetAsync(bnacc, 0, 128 * sizeof(float), stream);
  bn_stats<64><<<512, 256, 0, stream>>>(h2, NN, bnacc);
  bn_apply_elu<64><<<nblk(64 * N, BS), BS, 0, stream>>>(h2, NN, bnacc, g2, be2);

  // ---- pool + head ----
  hipMemsetAsync(pool, 0, GG * 64 * sizeof(float), stream);
  hipMemsetAsync(gcnt, 0, GG * sizeof(float), stream);
  pool_kernel<64><<<nblk(64 * N, BS), BS, 0, stream>>>(h2, batch, pool, gcnt);
  head_fc1<<<nblk(GG * 32, BS), BS, 0, stream>>>(pool, gcnt, fc1w, fc1b, z);
  hipMemsetAsync(bnacc, 0, 64 * sizeof(float), stream);
  bn_stats<32><<<64, 256, 0, stream>>>(z, GG, bnacc);
  head_out<<<nblk(GG, BS), BS, 0, stream>>>(z, bnacc, g3, be3, fc2w, fc2b, out);
}
